// TransformerBlock_64665027608967
// MI455X (gfx1250) — compile-verified, matrix operand order checked by reading, not run on hardware
//
#include <hip/hip_runtime.h>
#include <stdint.h>

// ---------------- CDNA5 WMMA plumbing (gfx1250, wave32) ----------------
typedef __attribute__((ext_vector_type(16))) __bf16 bf16x16;
typedef __attribute__((ext_vector_type(8)))  float  f32x8;

union FragU { unsigned u[8]; bf16x16 v; };

__device__ __forceinline__ unsigned short f2bfbits(float f) {
  union { float f; unsigned u; } v; v.f = f;
  unsigned r = v.u + 0x7FFFu + ((v.u >> 16) & 1u);   // round-to-nearest-even
  return (unsigned short)(r >> 16);
}

#if __has_builtin(__builtin_amdgcn_cvt_pk_bf16_f32)
typedef __attribute__((ext_vector_type(2))) __bf16 bf16x2;
__device__ __forceinline__ unsigned pack2bf(float lo, float hi) {
  bf16x2 p = __builtin_amdgcn_cvt_pk_bf16_f32(lo, hi);
  unsigned r; __builtin_memcpy(&r, &p, 4); return r;
}
#else
__device__ __forceinline__ unsigned pack2bf(float lo, float hi) {
  return (unsigned)f2bfbits(lo) | ((unsigned)f2bfbits(hi) << 16);
}
#endif

__device__ __forceinline__ f32x8 wmma_bf16(bf16x16 a, bf16x16 b, f32x8 c) {
  // (neg_a, A, neg_b, B, c_mod, C, reuse_a, reuse_b)
  return __builtin_amdgcn_wmma_f32_16x16x32_bf16(false, a, false, b, (short)0, c,
                                                 false, false);
}

__device__ __forceinline__ float gelu_tanh(float x) {
  float x3 = x * x * x;
  return 0.5f * x * (1.f + tanhf(0.7978845608028654f * (x + 0.044715f * x3)));
}

// Fragment layout (ISA 7.12.2, 16-bit A 16x32): lane half (lane>=16) owns K
// offsets +8; VGPR r<4 holds K=2r,2r+1(+8*half), r>=4 holds K=16+2(r-4)(+8*half).
// With K-pairs packed in uints a fragment is uints [4h..4h+3] and [8+4h..11+4h]
// of a row -> two aligned 16-byte LDS loads (row stride 20 uints = 80B keeps
// rows 16B-aligned and conflict-free for the 16 row-readers of a column).
__device__ __forceinline__ void frag_from_row(FragU& f, const unsigned* row, int half) {
  *(uint4*)&f.u[0] = *(const uint4*)(row + 4 * half);
  *(uint4*)&f.u[4] = *(const uint4*)(row + 8 + 4 * half);
}

// ---------------- Generic WMMA GEMM: C(MxN) = A(MxK,f32) * B(KxN,f32) ----------------
// MT = 64 or 128 rows per block (M%MT==0, N%64==0, K%32==0 for all calls here).
// MODE 0: C = acc + bias
// MODE 1: C = gelu(acc + bias)
// MODE 2: C = Res + acc + bias
// MODE 3: C += gate[m*gstride] * (acc + bias)   (MoE gated accumulate)
template<int MT, int MODE>
__global__ __launch_bounds__(128) void gemm_kernel(
    const float* __restrict__ A, const float* __restrict__ Bw,
    const float* __restrict__ bias, const float* __restrict__ Res,
    const float* __restrict__ gate, int gstride,
    float* __restrict__ C, int M, int N, int K)
{
  constexpr int RG = MT / 64;          // 16-row groups per wave
  __shared__ unsigned Asu[MT][20];     // [m][k/2] pair-packed bf16
  __shared__ unsigned Bsu[64][20];     // [n][k/2] (transposed during pack)
  const int tid  = threadIdx.x;
  const int wave = tid >> 5, lane = tid & 31;
  const int half = lane >> 4, nl = lane & 15;
  const int tileM = blockIdx.y * MT, tileN = blockIdx.x * 64;

  f32x8 acc[RG][4] = {};

  const int am = tid >> 3, ak = (tid & 7) * 4;   // A stage coords

  for (int k0 = 0; k0 < K; k0 += 32) {
    __syncthreads();
    // ---- stage A MTx32: float4 loads, pair-packed uint2 stores ----
#pragma unroll
    for (int it = 0; it < MT / 16; ++it) {
      int m = am + 16 * it;
      float4 va = *(const float4*)(A + (size_t)(tileM + m) * K + k0 + ak);
      uint2 p; p.x = pack2bf(va.x, va.y); p.y = pack2bf(va.z, va.w);
      *(uint2*)&Asu[m][ak >> 1] = p;
    }
    // ---- stage B 32x64: coalesced row pairs, transpose via pair-packing ----
#pragma unroll
    for (int it = 0; it < 2; ++it) {
      int idx = tid + 128 * it;            // 0..255
      int kp = idx >> 4;                   // k pair 0..15
      int n4 = (idx & 15) << 2;            // 0,4,..,60
      const float* b0 = Bw + (size_t)(k0 + 2 * kp) * N + tileN + n4;
      float4 r0 = *(const float4*)b0;
      float4 r1 = *(const float4*)(b0 + N);
      Bsu[n4 + 0][kp] = pack2bf(r0.x, r1.x);
      Bsu[n4 + 1][kp] = pack2bf(r0.y, r1.y);
      Bsu[n4 + 2][kp] = pack2bf(r0.z, r1.z);
      Bsu[n4 + 3][kp] = pack2bf(r0.w, r1.w);
    }
    __syncthreads();

    FragU fa[RG];
#pragma unroll
    for (int g = 0; g < RG; ++g)
      frag_from_row(fa[g], &Asu[64 * g + 16 * wave + nl][0], half);
#pragma unroll
    for (int nt = 0; nt < 4; ++nt) {
      FragU fb;
      frag_from_row(fb, &Bsu[nt * 16 + nl][0], half);
#pragma unroll
      for (int g = 0; g < RG; ++g)
        acc[g][nt] = wmma_bf16(fa[g].v, fb.v, acc[g][nt]);
    }
  }

  // epilogue (C layout: VGPR v -> M = v + 8*half, N = lane&15)
#pragma unroll
  for (int g = 0; g < RG; ++g) {
    const int mbase = tileM + 64 * g + 16 * wave + 8 * half;
#pragma unroll
    for (int nt = 0; nt < 4; ++nt) {
      int col = tileN + nt * 16 + nl;
      float bv = bias[col];
#pragma unroll
      for (int v = 0; v < 8; ++v) {
        int m = mbase + v;
        float val = acc[g][nt][v] + bv;
        size_t idx = (size_t)m * N + col;
        if (MODE == 1) val = gelu_tanh(val);
        if (MODE == 2) val += Res[idx];
        if (MODE == 3) C[idx] += gate[(size_t)m * gstride] * val;
        else           C[idx] = val;
      }
    }
  }
}

// build a Q/K-style fragment pair (two K=32 chunks over HD=64) from one
// contiguous 64-float row: chunk q covers hd [16q+8*half, +8).
__device__ __forceinline__ void qk_frags(FragU f[2], const float* row8h, bool valid,
                                         float scale) {
#pragma unroll
  for (int q = 0; q < 4; ++q) {
    float4 a = *(const float4*)(row8h + 16 * q);
    float4 c = *(const float4*)(row8h + 16 * q + 4);
    if (!valid) { a.x = a.y = a.z = a.w = 0.f; c.x = c.y = c.z = c.w = 0.f; }
    unsigned* up = f[q >> 1].u + 4 * (q & 1);
    up[0] = pack2bf(a.x * scale, a.y * scale);
    up[1] = pack2bf(a.z * scale, a.w * scale);
    up[2] = pack2bf(c.x * scale, c.y * scale);
    up[3] = pack2bf(c.z * scale, c.w * scale);
  }
}

// ---------------- Flash attention (1 wave = 16 queries x HD=64) ----------------
__global__ __launch_bounds__(32) void flash_kernel(
    const float* __restrict__ Q, const float* __restrict__ Kb,
    const float* __restrict__ Vb, const unsigned char* __restrict__ kpm,
    float* __restrict__ O, int Sq, int Sk, float scale)
{
  const int D = 512, HD = 64;
  const int qt = blockIdx.x, head = blockIdx.y, b = blockIdx.z;
  const int lane = threadIdx.x;
  const int half = lane >> 4, nl = lane & 15;
  __shared__ unsigned Vsu[64][20];   // [hd][key/2] pair-packed bf16
  __shared__ unsigned Pu[16][16];    // [q][key/2]  pair-packed bf16

  FragU aq[2];
  qk_frags(aq, Q + ((size_t)(b * Sq + qt * 16 + nl)) * D + head * HD + 8 * half,
           true, scale);

  f32x8 oacc[4] = {};
  float rmax[8], rsum[8];
#pragma unroll
  for (int v = 0; v < 8; ++v) { rmax[v] = -1e30f; rsum[v] = 0.f; }

  const int nkb = (Sk + 31) / 32;
  for (int kblk = 0; kblk < nkb; ++kblk) {
    // ---- stage V 32x64 into LDS (transpose via pair packing) ----
    __syncthreads();
#pragma unroll
    for (int it = 0; it < 8; ++it) {
      int idx = lane + 32 * it;            // 0..255
      int kp = idx >> 4;                   // key pair 0..15
      int n4 = (idx & 15) << 2;
      int kA = kblk * 32 + 2 * kp, kB = kA + 1;
      float4 r0 = *(const float4*)(Vb + ((size_t)(b * Sk + (kA < Sk ? kA : 0))) * D
                                   + head * HD + n4);
      float4 r1 = *(const float4*)(Vb + ((size_t)(b * Sk + (kB < Sk ? kB : 0))) * D
                                   + head * HD + n4);
      if (kA >= Sk) { r0.x = r0.y = r0.z = r0.w = 0.f; }
      if (kB >= Sk) { r1.x = r1.y = r1.z = r1.w = 0.f; }
      Vsu[n4 + 0][kp] = pack2bf(r0.x, r1.x);
      Vsu[n4 + 1][kp] = pack2bf(r0.y, r1.y);
      Vsu[n4 + 2][kp] = pack2bf(r0.z, r1.z);
      Vsu[n4 + 3][kp] = pack2bf(r0.w, r1.w);
    }

    // ---- scores: two 16x16 tiles (32 keys), K^T fragments from K rows ----
    f32x8 s0 = {}, s1 = {};
#pragma unroll
    for (int t = 0; t < 2; ++t) {
      int key = kblk * 32 + t * 16 + nl;
      bool kv = key < Sk;
      FragU fk[2];
      qk_frags(fk, Kb + ((size_t)(b * Sk + (kv ? key : 0))) * D + head * HD + 8 * half,
               kv, 1.f);
      if (t == 0) { s0 = wmma_bf16(aq[0].v, fk[0].v, s0); s0 = wmma_bf16(aq[1].v, fk[1].v, s0); }
      else        { s1 = wmma_bf16(aq[0].v, fk[0].v, s1); s1 = wmma_bf16(aq[1].v, fk[1].v, s1); }
    }

    // ---- masking + online softmax (rows m = v + 8*half) ----
    int key0 = kblk * 32 + nl, key1 = kblk * 32 + 16 + nl;
    bool in0 = key0 < Sk, in1 = key1 < Sk;
    bool mk0 = in0 && kpm && kpm[(size_t)b * Sk + key0];
    bool mk1 = in1 && kpm && kpm[(size_t)b * Sk + key1];
    float sm0[8], sm1[8], lm[8];
#pragma unroll
    for (int v = 0; v < 8; ++v) {
      sm0[v] = !in0 ? -1e30f : (mk0 ? -1e9f : s0[v]);
      sm1[v] = !in1 ? -1e30f : (mk1 ? -1e9f : s1[v]);
      lm[v]  = fmaxf(sm0[v], sm1[v]);
    }
    for (int off = 1; off < 16; off <<= 1)
#pragma unroll
      for (int v = 0; v < 8; ++v)
        lm[v] = fmaxf(lm[v], __shfl_xor(lm[v], off, 32));
    float alpha[8], p0[8], p1[8], lsum[8];
#pragma unroll
    for (int v = 0; v < 8; ++v) {
      float nm = fmaxf(rmax[v], lm[v]);
      alpha[v] = __expf(rmax[v] - nm);
      rmax[v] = nm;
      p0[v] = __expf(sm0[v] - nm);
      p1[v] = __expf(sm1[v] - nm);
      lsum[v] = p0[v] + p1[v];
    }
    for (int off = 1; off < 16; off <<= 1)
#pragma unroll
      for (int v = 0; v < 8; ++v)
        lsum[v] += __shfl_xor(lsum[v], off, 32);
#pragma unroll
    for (int v = 0; v < 8; ++v) rsum[v] = rsum[v] * alpha[v] + lsum[v];
#pragma unroll
    for (int nt = 0; nt < 4; ++nt)
#pragma unroll
      for (int v = 0; v < 8; ++v) oacc[nt][v] = oacc[nt][v] * alpha[v];

    // ---- P (C-layout) -> pair-packed LDS -> A-fragment layout ----
#pragma unroll
    for (int v = 0; v < 8; ++v) {
      unsigned short* prow = (unsigned short*)&Pu[v + 8 * half][0];
      prow[nl]      = f2bfbits(p0[v]);
      prow[16 + nl] = f2bfbits(p1[v]);
    }
    __syncthreads();
    FragU fp;
    frag_from_row(fp, &Pu[nl][0], half);

    // ---- PV: contraction over 32 keys, N = 64 hd cols ----
#pragma unroll
    for (int nt = 0; nt < 4; ++nt) {
      FragU fv;
      frag_from_row(fv, &Vsu[nt * 16 + nl][0], half);
      oacc[nt] = wmma_bf16(fp.v, fv.v, oacc[nt]);
    }
  }

#pragma unroll
  for (int nt = 0; nt < 4; ++nt)
#pragma unroll
    for (int v = 0; v < 8; ++v) {
      int m = qt * 16 + v + 8 * half;
      O[((size_t)(b * Sq + m)) * D + head * HD + nt * 16 + nl] = oacc[nt][v] / rsum[v];
    }
}

// ---------------- LN / AdaLN / modulation / gating ----------------
__global__ __launch_bounds__(256) void mod_kernel(
    const float* __restrict__ emb, const float* __restrict__ w,
    const float* __restrict__ bias, float* __restrict__ out, int twoD)
{
  int idx = blockIdx.x * 256 + threadIdx.x;
  int b = idx / twoD, j = idx % twoD;
  const float* e = emb + (size_t)b * 512;
  float acc = bias[j];
  for (int k = 0; k < 512; ++k) {
    float ev = e[k];
    acc += (ev / (1.f + __expf(-ev))) * w[(size_t)k * twoD + j];
  }
  out[idx] = acc;
}

__global__ __launch_bounds__(256) void adaln_kernel(
    const float* __restrict__ X, const float* __restrict__ mod,
    float* __restrict__ out, int S)
{
  int t = blockIdx.x, b = t / S, tid = threadIdx.x;
  __shared__ float red[256];
  float x0 = X[(size_t)t * 512 + tid];
  float x1 = X[(size_t)t * 512 + 256 + tid];
  red[tid] = x0 + x1; __syncthreads();
  for (int o = 128; o > 0; o >>= 1) { if (tid < o) red[tid] += red[tid + o]; __syncthreads(); }
  float mean = red[0] * (1.f / 512.f); __syncthreads();
  float d0 = x0 - mean, d1 = x1 - mean;
  red[tid] = d0 * d0 + d1 * d1; __syncthreads();
  for (int o = 128; o > 0; o >>= 1) { if (tid < o) red[tid] += red[tid + o]; __syncthreads(); }
  float rstd = rsqrtf(red[0] * (1.f / 512.f) + 1e-5f);
  const float* mr = mod + (size_t)b * 1024;
  out[(size_t)t * 512 + tid]       = d0 * rstd * (1.f + mr[tid])       + mr[512 + tid];
  out[(size_t)t * 512 + 256 + tid] = d1 * rstd * (1.f + mr[256 + tid]) + mr[768 + tid];
}

__global__ __launch_bounds__(256) void lny_kernel(
    const float* __restrict__ Y, const float* __restrict__ g,
    const float* __restrict__ bb, float* __restrict__ out)
{
  int t = blockIdx.x, tid = threadIdx.x;
  __shared__ float red[256];
  float x0 = Y[(size_t)t * 512 + tid];
  float x1 = Y[(size_t)t * 512 + 256 + tid];
  red[tid] = x0 + x1; __syncthreads();
  for (int o = 128; o > 0; o >>= 1) { if (tid < o) red[tid] += red[tid + o]; __syncthreads(); }
  float mean = red[0] * (1.f / 512.f); __syncthreads();
  float d0 = x0 - mean, d1 = x1 - mean;
  red[tid] = d0 * d0 + d1 * d1; __syncthreads();
  for (int o = 128; o > 0; o >>= 1) { if (tid < o) red[tid] += red[tid + o]; __syncthreads(); }
  float rstd = rsqrtf(red[0] * (1.f / 512.f) + 1e-5f);
  out[(size_t)t * 512 + tid]       = d0 * rstd * g[tid]       + bb[tid];
  out[(size_t)t * 512 + 256 + tid] = d1 * rstd * g[256 + tid] + bb[256 + tid];
}

__global__ __launch_bounds__(256) void logits_kernel(
    const float* __restrict__ xf, const float* __restrict__ emb,
    const float* __restrict__ gw, const float* __restrict__ tgw,
    float* __restrict__ logits)
{
  int idx = blockIdx.x * 256 + threadIdx.x;   // < 16384*8
  int s = idx >> 3, e = idx & 7, b = s >> 8;  // S=256 tokens per batch
  const float* xr = xf + (size_t)s * 512;
  const float* g  = gw + (size_t)e * 512;
  const float* er = emb + (size_t)b * 512;
  const float* tg = tgw + (size_t)e * 512;
  float acc = 0.f;
  for (int d = 0; d < 512; ++d) acc += xr[d] * g[d] + er[d] * tg[d];
  logits[idx] = acc;
}

__global__ __launch_bounds__(256) void gate_kernel(
    const float* __restrict__ logits, float* __restrict__ gate)
{
  int s = blockIdx.x * 256 + threadIdx.x;
  float l[8], p[8], mx = -1e30f, sum = 0.f;
#pragma unroll
  for (int e = 0; e < 8; ++e) { l[e] = logits[(size_t)s * 8 + e]; mx = fmaxf(mx, l[e]); }
#pragma unroll
  for (int e = 0; e < 8; ++e) { p[e] = __expf(l[e] - mx); sum += p[e]; }
  float inv = 1.f / sum;
#pragma unroll
  for (int e = 0; e < 8; ++e)
    gate[(size_t)s * 8 + e] = (l[e] > 0.f) ? p[e] * inv : 0.f;  // sigmoid(l)-0.5>0 <=> l>0
}

// ---------------- Orchestration ----------------
extern "C" void kernel_launch(void* const* d_in, const int* in_sizes, int n_in,
                              void* d_out, int out_size, void* d_ws, size_t ws_size,
                              hipStream_t stream)
{
  (void)in_sizes; (void)n_in; (void)out_size; (void)ws_size;
  const float* x   = (const float*)d_in[0];
  const float* y   = (const float*)d_in[1];
  const float* emb = (const float*)d_in[2];
  const unsigned char* kpm = (const unsigned char*)d_in[3];
  const float* sa_mod_w = (const float*)d_in[4];  const float* sa_mod_b = (const float*)d_in[5];
  const float* sa_wq = (const float*)d_in[6];     const float* sa_bq = (const float*)d_in[7];
  const float* sa_wk = (const float*)d_in[8];     const float* sa_bk = (const float*)d_in[9];
  const float* sa_wv = (const float*)d_in[10];    const float* sa_bv = (const float*)d_in[11];
  const float* sa_wo = (const float*)d_in[12];    const float* sa_bo = (const float*)d_in[13];
  const float* ca_mod_w = (const float*)d_in[14]; const float* ca_mod_b = (const float*)d_in[15];
  const float* ca_ln_g = (const float*)d_in[16];  const float* ca_ln_b = (const float*)d_in[17];
  const float* ca_wq = (const float*)d_in[18];    const float* ca_bq = (const float*)d_in[19];
  const float* ca_wk = (const float*)d_in[20];    const float* ca_bk = (const float*)d_in[21];
  const float* ca_wv = (const float*)d_in[22];    const float* ca_bv = (const float*)d_in[23];
  const float* ca_wo = (const float*)d_in[24];    const float* ca_bo = (const float*)d_in[25];
  const float* moe_mod_w = (const float*)d_in[26];const float* moe_mod_b = (const float*)d_in[27];
  const float* gate_w = (const float*)d_in[28];   const float* text_gate_w = (const float*)d_in[29];
  const float* exp_w1 = (const float*)d_in[30];   const float* exp_b1 = (const float*)d_in[31];
  const float* exp_w2 = (const float*)d_in[32];   const float* exp_b2 = (const float*)d_in[33];

  char* ws = (char*)d_ws;
  float* mod_sa = (float*)(ws + 0x000000);     // 64x1024 f32
  float* mod_ca = (float*)(ws + 0x040000);
  float* mod_mo = (float*)(ws + 0x080000);
  float* logits = (float*)(ws + 0x0C0000);     // 16384x8
  float* gatev  = (float*)(ws + 0x140000);     // 16384x8
  float* hbuf = (float*)(ws + 0x0200000);      // 16384x512 f32 (h / hq / xf)
  float* qb   = (float*)(ws + 0x2200000);      // 16384x512
  float* kb   = (float*)(ws + 0x4200000);
  float* vb   = (float*)(ws + 0x6200000);
  float* attn = (float*)(ws + 0x8200000);
  float* h1   = (float*)(ws + 0xA200000);
  float* h2   = (float*)(ws + 0xC200000);
  float* yn   = (float*)(ws + 0xE200000);      // 4928x512
  float* hmid = qb;                            // aliases qb+kb (64 MiB) during MoE
  float* outf = (float*)d_out;

  const dim3 gBig(8, 128);    // N=512,  M=16384, MT=128
  const dim3 gYn(8, 77);      // N=512,  M=4928,  MT=64
  const dim3 gF1(16, 128);    // N=1024, M=16384, MT=128
  const dim3 gf(16, 8, 64);   // q-tiles, heads, batch
  const float scale = 0.125f; // 1/sqrt(64)

  // AdaLN modulation vectors
  mod_kernel<<<256, 256, 0, stream>>>(emb, sa_mod_w,  sa_mod_b,  mod_sa, 1024);
  mod_kernel<<<256, 256, 0, stream>>>(emb, ca_mod_w,  ca_mod_b,  mod_ca, 1024);
  mod_kernel<<<256, 256, 0, stream>>>(emb, moe_mod_w, moe_mod_b, mod_mo, 1024);

  // ---- self attention ----
  adaln_kernel<<<16384, 256, 0, stream>>>(x, mod_sa, hbuf, 256);
  gemm_kernel<128,0><<<gBig, 128, 0, stream>>>(hbuf, sa_wq, sa_bq, nullptr, nullptr, 0, qb, 16384, 512, 512);
  gemm_kernel<128,0><<<gBig, 128, 0, stream>>>(hbuf, sa_wk, sa_bk, nullptr, nullptr, 0, kb, 16384, 512, 512);
  gemm_kernel<128,0><<<gBig, 128, 0, stream>>>(hbuf, sa_wv, sa_bv, nullptr, nullptr, 0, vb, 16384, 512, 512);
  flash_kernel<<<gf, 32, 0, stream>>>(qb, kb, vb, kpm, attn, 256, 256, scale);
  gemm_kernel<128,2><<<gBig, 128, 0, stream>>>(attn, sa_wo, sa_bo, x, nullptr, 0, h1, 16384, 512, 512);

  // ---- cross attention ----
  lny_kernel<<<4928, 256, 0, stream>>>(y, ca_ln_g, ca_ln_b, yn);
  adaln_kernel<<<16384, 256, 0, stream>>>(h1, mod_ca, hbuf, 256);
  gemm_kernel<128,0><<<gBig, 128, 0, stream>>>(hbuf, ca_wq, ca_bq, nullptr, nullptr, 0, qb, 16384, 512, 512);
  gemm_kernel<64,0><<<gYn, 128, 0, stream>>>(yn, ca_wk, ca_bk, nullptr, nullptr, 0, kb, 4928, 512, 512);
  gemm_kernel<64,0><<<gYn, 128, 0, stream>>>(yn, ca_wv, ca_bv, nullptr, nullptr, 0, vb, 4928, 512, 512);
  flash_kernel<<<gf, 32, 0, stream>>>(qb, kb, vb, nullptr, attn, 256, 77, scale);
  gemm_kernel<128,2><<<gBig, 128, 0, stream>>>(attn, ca_wo, ca_bo, h1, nullptr, 0, h2, 16384, 512, 512);

  // ---- MoE FFN ----
  adaln_kernel<<<16384, 256, 0, stream>>>(h2, mod_mo, hbuf, 256);
  logits_kernel<<<512, 256, 0, stream>>>(hbuf, emb, gate_w, text_gate_w, logits);
  gate_kernel<<<64, 256, 0, stream>>>(logits, gatev);
  hipMemcpyAsync(d_out, h2, (size_t)16384 * 512 * sizeof(float),
                 hipMemcpyDeviceToDevice, stream);
  for (int e = 0; e < 8; ++e) {
    gemm_kernel<128,1><<<gF1, 128, 0, stream>>>(
        hbuf, exp_w1 + (size_t)e * 512 * 1024, exp_b1 + (size_t)e * 1024,
        nullptr, nullptr, 0, hmid, 16384, 1024, 512);
    gemm_kernel<128,3><<<gBig, 128, 0, stream>>>(
        hmid, exp_w2 + (size_t)e * 1024 * 512, exp_b2 + (size_t)e * 512,
        nullptr, gatev + e, 8, outf, 16384, 512, 1024);
  }
}
